// NMI_43104291783207
// MI455X (gfx1250) — matrix operand order repaired and checked by reading, not compiled, not run on hardware
//
#include <hip/hip_runtime.h>

typedef __attribute__((ext_vector_type(16))) _Float16 v16h;
typedef __attribute__((ext_vector_type(2)))  __fp16   hp2;   // cvt_pkrtz result type
typedef __attribute__((ext_vector_type(8)))  float    v8f;
typedef __attribute__((ext_vector_type(2)))  float    f2;

#define NBINS   16
#define C2      16230.32f    // PRETERM * log2(e) = 11250 * 1.4426950409
#define BINSTEP (1.0f/15.0f)
#define EPSF    1e-6f
#define WAVES_PER_BLOCK 8
#define VPW 64               // voxels per wave-iteration (2 WMMA K-slabs)

union Frag16 { v16h h; uint4 q[2]; };
union Pack2  { hp2 h; unsigned int u; };

// ---------------------------------------------------------------------------
// Zero the 2 x 256 f32 joint-histogram workspace.
// ---------------------------------------------------------------------------
__global__ void nmi_zero_kernel(float* hist) {
    int i = blockIdx.x * blockDim.x + threadIdx.x;
    if (i < 2 * 256) hist[i] = 0.0f;
}

// ---------------------------------------------------------------------------
// Main kernel: per-voxel soft histogram (VALU, exp2-based) feeding the
// 16x16xV joint-histogram GEMM via v_wmma_f32_16x16x32_f16.
// ---------------------------------------------------------------------------
__global__ void __launch_bounds__(256)
nmi_joint_kernel(const float* __restrict__ y_true,
                 const float* __restrict__ y_pred,
                 float* __restrict__ hist,
                 int V /* voxels per batch */) {
    const int lane  = threadIdx.x & 31;
    const int wv    = threadIdx.x >> 5;             // wave in block, 0..7
    const int batch = blockIdx.y;

    // Per-wave staging: [image][bin][voxel-in-64-slab] f16, bin-major so
    // WMMA fragment loads are contiguous ds_load_b128.
    __shared__ __align__(16) _Float16 sm[WAVES_PER_BLOCK][2][NBINS * VPW];
    // Cross-wave partial-histogram reduction buffer.
    __shared__ float acc[WAVES_PER_BLOCK][256];

    const float* ya = y_true + (size_t)batch * V;
    const float* yb = y_pred + (size_t)batch * V;

    const int nSlabs = V / VPW;                     // 32768 for V = 128^3
    const int wid    = blockIdx.x * WAVES_PER_BLOCK + wv;
    const int wtot   = gridDim.x * WAVES_PER_BLOCK;

    _Float16* At = &sm[wv][0][0];
    _Float16* Bt = &sm[wv][1][0];

    const int m  = lane & 15;                       // row (A) / col (B) index
    const int kg = (lane < 16) ? 0 : 8;             // A-matrix K group base
    const int kb = (lane < 16) ? 0 : 16;            // unused; B loads use kg like A

    v8f c0 = {}, c1 = {};                           // independent accumulators

    for (int t = wid; t < nSlabs; t += wtot) {
        // Each lane owns two consecutive voxels of the 64-voxel slab.
        const int v0 = t * VPW + 2 * lane;
        const f2 xa = __builtin_nontemporal_load((const f2*)(ya + v0));
        const f2 xb = __builtin_nontemporal_load((const f2*)(yb + v0));

        // --- soft histogram weights for both voxels, both images ----------
        float wa0[NBINS], wa1[NBINS], wb0[NBINS], wb1[NBINS];
        float sa0 = 0.f, sa1 = 0.f, sb0 = 0.f, sb1 = 0.f;
#pragma unroll
        for (int i = 0; i < NBINS; ++i) {
            const float ci = (float)i * BINSTEP;
            float d;
            d = xa.x - ci; wa0[i] = exp2f(-C2 * d * d); sa0 += wa0[i];
            d = xa.y - ci; wa1[i] = exp2f(-C2 * d * d); sa1 += wa1[i];
            d = xb.x - ci; wb0[i] = exp2f(-C2 * d * d); sb0 += wb0[i];
            d = xb.y - ci; wb1[i] = exp2f(-C2 * d * d); sb1 += wb1[i];
        }
        const float ia0 = 1.0f / sa0, ia1 = 1.0f / sa1;
        const float ib0 = 1.0f / sb0, ib1 = 1.0f / sb1;

        // Pack the voxel pair with v_cvt_pk_rtz_f16_f32, one b32 store/bin.
#pragma unroll
        for (int i = 0; i < NBINS; ++i) {
            Pack2 pa, pb;
            pa.h = __builtin_amdgcn_cvt_pkrtz(wa0[i] * ia0, wa1[i] * ia1);
            pb.h = __builtin_amdgcn_cvt_pkrtz(wb0[i] * ib0, wb1[i] * ib1);
            *(unsigned int*)(At + i * VPW + 2 * lane) = pa.u;
            *(unsigned int*)(Bt + i * VPW + 2 * lane) = pb.u;
        }
        __builtin_amdgcn_wave_barrier();            // keep DS stores before loads

        // --- two K=32 WMMA steps over the 64-voxel slab -------------------
#pragma unroll
        for (int s = 0; s < 2; ++s) {
            Frag16 fa, fb;
            const int ks = 32 * s;
            // A (16x32 f16): lane m holds K = {kg..kg+7, kg+16..kg+23}
            fa.q[0] = *(const uint4*)(At + m * VPW + ks + kg);
            fa.q[1] = *(const uint4*)(At + m * VPW + ks + kg + 16);
            // B (32x16 f16): lane holds column n=m, K = {kg..kg+7, kg+16..kg+23}
            fb.q[0] = *(const uint4*)(Bt + m * VPW + ks + kg);
            fb.q[1] = *(const uint4*)(Bt + m * VPW + ks + kg + 16);
            if (s == 0)
                c0 = __builtin_amdgcn_wmma_f32_16x16x32_f16(
                        false, fa.h, false, fb.h, (short)0, c0, false, false);
            else
                c1 = __builtin_amdgcn_wmma_f32_16x16x32_f16(
                        false, fa.h, false, fb.h, (short)0, c1, false, false);
        }
    }

    // --- reduce the 8 per-wave partial 16x16 tiles inside the block -------
    // C/D layout: element r of lane l -> row (r + (l<16?0:8)), col (l&15)
    const int row0 = (lane < 16) ? 0 : 8;
#pragma unroll
    for (int r = 0; r < 8; ++r)
        acc[wv][(r + row0) * 16 + m] = c0[r] + c1[r];
    __syncthreads();

    if (threadIdx.x < 256) {
        float s = 0.0f;
#pragma unroll
        for (int w = 0; w < WAVES_PER_BLOCK; ++w) s += acc[w][threadIdx.x];
        atomicAdd(&hist[batch * 256 + threadIdx.x], s);
    }
}

// ---------------------------------------------------------------------------
// Finalize: entropies + NMI loss, one block per batch.
// ---------------------------------------------------------------------------
__global__ void __launch_bounds__(256)
nmi_finalize_kernel(const float* __restrict__ hist,
                    float* __restrict__ out, float invV) {
    const int b = blockIdx.x;
    const int t = threadIdx.x;

    __shared__ float P[256];
    __shared__ float red[256];
    __shared__ float mrg[32];
    __shared__ float HxyS;

    const float p = hist[b * 256 + t] * invV;
    P[t]   = p;
    red[t] = -p * __log2f(p + EPSF);
    __syncthreads();

    // block reduction -> Hxy
    for (int s = 128; s > 0; s >>= 1) {
        if (t < s) red[t] += red[t + s];
        __syncthreads();
    }
    if (t == 0) HxyS = red[0];
    __syncthreads();

    // marginals: pa = rowsum(pab), pb = colsum(pab)
    if (t < 16) {
        float pa = 0.0f;
#pragma unroll
        for (int j = 0; j < 16; ++j) pa += P[t * 16 + j];
        mrg[t] = -pa * __log2f(pa + EPSF);
    } else if (t < 32) {
        const int n = t - 16;
        float pb = 0.0f;
#pragma unroll
        for (int i = 0; i < 16; ++i) pb += P[i * 16 + n];
        mrg[t] = -pb * __log2f(pb + EPSF);
    }
    __syncthreads();

    if (t == 0) {
        float Hx = 0.0f, Hy = 0.0f;
#pragma unroll
        for (int i = 0; i < 16; ++i) { Hx += mrg[i]; Hy += mrg[16 + i]; }
        const float nmi = 2.0f * (1.0f - HxyS / (Hx + Hy));
        out[b] = 1.0f - nmi;
    }
}

// ---------------------------------------------------------------------------
extern "C" void kernel_launch(void* const* d_in, const int* in_sizes, int n_in,
                              void* d_out, int out_size, void* d_ws, size_t ws_size,
                              hipStream_t stream) {
    const float* y_true = (const float*)d_in[0];
    const float* y_pred = (const float*)d_in[1];
    float* out  = (float*)d_out;
    float* hist = (float*)d_ws;                 // 2 x 256 f32

    const int V = in_sizes[0] / 2;              // voxels per batch = 128^3

    nmi_zero_kernel<<<2, 256, 0, stream>>>(hist);

    dim3 grid(256, 2, 1);                       // x: blocks, y: batch
    nmi_joint_kernel<<<grid, 256, 0, stream>>>(y_true, y_pred, hist, V);

    nmi_finalize_kernel<<<2, 256, 0, stream>>>(hist, out, 1.0f / (float)V);
}
